// MultiScaleDecoder_11373073400244
// MI455X (gfx1250) — matrix operand order repaired and checked by reading, not compiled, not run on hardware
//
#include <hip/hip_runtime.h>
#include <hip/hip_bf16.h>
#include <math.h>

// ---------------------------------------------------------------------------
// MultiScaleDecoder on gfx1250 (MI455X): persistent per-batch-tile kernel.
//   B=256, T=512, Z=128, E=256, HS=512, HF=1024, K=4
// One workgroup (256 threads = 8 waves, wave32) owns 16 batch rows and walks
// all 512 timesteps. All matmuls via v_wmma_f32_16x16x32_bf16.
// Software pipelining of the weight streams is enforced with
// __builtin_amdgcn_sched_group_barrier (WMMA/VMEM interleave spec) so the
// scheduler issues loads 2 iterations ahead instead of s_wait_loadcnt 0x0
// before every WMMA.
// ---------------------------------------------------------------------------

typedef __bf16 bf16;
typedef __attribute__((ext_vector_type(16))) __bf16 v16bf;
typedef __attribute__((ext_vector_type(8)))  float  v8f;

static_assert(sizeof(v16bf) == 32, "v16bf must be 32 bytes");

#define ZD  128
#define ED  256
#define HS  512
#define HF  1024
#define BB  256
#define TT  512

// ---- scheduler pipeline spec helpers ----
#if defined(__has_builtin)
#  if __has_builtin(__builtin_amdgcn_sched_group_barrier)
#    define SGB(mask, size) __builtin_amdgcn_sched_group_barrier((mask), (size), 0)
#  endif
#endif
#ifndef SGB
#  define SGB(mask, size)
#endif
#define SGB_MMA  0x008   // MFMA/WMMA
#define SGB_VMEM 0x020   // VMEM reads

// Interleave spec: PRE+2*LPI loads up front, then (MMA WPI / VMEM LPI) pairs,
// tail of 2*WPI MMAs -> loads run 2 iterations ahead of their consumers.
template <int ITERS, int LPI, int WPI, int PRE>
__device__ __forceinline__ void sched_interleave() {
    SGB(SGB_VMEM, PRE + 2 * LPI);
#pragma unroll
    for (int i = 0; i < ITERS - 2; ++i) {
        SGB(SGB_MMA, WPI);
        SGB(SGB_VMEM, LPI);
    }
    SGB(SGB_MMA, 2 * WPI);
}

// ---- workspace layout (bytes) ----
#define O_WHH  ((size_t)0)                                // fast_w_hh  3072x1024
#define O_WZ   (O_WHH + (size_t)3072*1024*2)              // fast_w_ih[:, :128]
#define O_WH   (O_WZ  + (size_t)3072*128*2)               // fast_w_ih[:, 128:640]
#define O_SHH  (O_WH  + (size_t)3072*512*2)               // slow_w_hh  1536x512
#define O_MUW  (O_SHH + (size_t)1536*512*2)               // mu_w 128x1024
#define O_LVW  (O_MUW + (size_t)128*1024*2)               // lv_w 128x1024
#define O_GIS  (O_LVW + (size_t)128*1024*2)               // gi_slow f32 [256][1536]
#define O_GIHS (O_GIS + (size_t)256*1536*4)               // per-block gihs f32 [16][3][16][1024]

// ---------------------------------------------------------------------------
// Fragment helpers (layouts per ISA 7.12.2, packed per-lane-contiguous).
// ---------------------------------------------------------------------------
__device__ __forceinline__ v16bf load_frag(const bf16* p) {
    union { uint4 q[2]; v16bf v; } u;
    const uint4* s = (const uint4*)p;
    u.q[0] = s[0];
    u.q[1] = s[1];
    return u.v;
}

__device__ __forceinline__ v8f wmma_bf16(v16bf a, v16bf b, v8f c) {
    return __builtin_amdgcn_wmma_f32_16x16x32_bf16(false, a, false, b, (short)0, c,
                                                   false, false);
}

__device__ __forceinline__ const bf16* btile(const bf16* base, int ntile, int nkt,
                                             int kt, int lane) {
    return base + (((size_t)(ntile * nkt + kt) * 32 + lane) << 4);
}

struct B3 { v16bf r, z, h; };

__device__ __forceinline__ B3 loadB3(const bf16* base, int ntR, int ntZ, int ntH,
                                     int nkt, int kt, int lane) {
    B3 b;
    b.r = load_frag(btile(base, ntR, nkt, kt, lane));
    b.z = load_frag(btile(base, ntZ, nkt, kt, lane));
    b.h = load_frag(btile(base, ntH, nkt, kt, lane));
    return b;
}

__device__ __forceinline__ float sigf(float x) { return 1.0f / (1.0f + __expf(-x)); }

// Repack a row-major [16][Kdim] bf16 LDS buffer into A-fragment layout.
__device__ __forceinline__ void packA_lds(const bf16* rm, bf16* A, int Kdim, int tid) {
    int slots = Kdim;  // (Kdim/32 tiles) * 32 lanes
    for (int s = tid; s < slots; s += 256) {
        int kt = s >> 5, L = s & 31;
        int M = L & 15, hf = L >> 4;
        const uint4* s0 = (const uint4*)(rm + (size_t)M * Kdim + kt * 32 + 8 * hf);
        const uint4* s1 = (const uint4*)(rm + (size_t)M * Kdim + kt * 32 + 16 + 8 * hf);
        uint4* d = (uint4*)(A + ((size_t)(kt * 32 + L) << 4));
        d[0] = *s0;
        d[1] = *s1;
    }
}

// gihs[g][M][j] = b_ih[g*HF+j] (+ b_hh for r,z) + sum_k h_slow[M,k]*Wih[g*HF+j, 128+k]
__device__ __forceinline__ void compute_gihs(const bf16* hsA, const bf16* pWh,
                                             const float* __restrict__ fbih,
                                             const float* __restrict__ fbhh,
                                             float* __restrict__ gihs,
                                             int wave, int lane, int halfid, int ncol) {
#pragma unroll 1
    for (int g = 0; g < 3; ++g) {
#pragma unroll 1
        for (int ct = 0; ct < 8; ++ct) {
            int j0 = wave * 128 + ct * 16;
            int jc = j0 + ncol;
            float binit = fbih[g * HF + jc] + ((g < 2) ? fbhh[g * HF + jc] : 0.0f);
            v8f acc;
#pragma unroll
            for (int v = 0; v < 8; ++v) acc[v] = binit;
            int nt = g * 64 + (j0 >> 4);
#pragma unroll
            for (int kt = 0; kt < 16; ++kt) {
                v16bf a = load_frag(hsA + ((size_t)(kt * 32 + lane) << 4));
                acc = wmma_bf16(a, load_frag(btile(pWh, nt, 16, kt, lane)), acc);
            }
            sched_interleave<16, 2, 1, 2>();
#pragma unroll
            for (int v = 0; v < 8; ++v)
                gihs[(size_t)(g * 16 + (v + 8 * halfid)) * HF + jc] = acc[v];
        }
    }
}

// ---------------------------------------------------------------------------
// Prep kernels
// ---------------------------------------------------------------------------
__global__ void pack_bfrag_kernel(const float* __restrict__ W, bf16* __restrict__ dst,
                                  int Kdim, int ld, int coloff) {
    int idx = blockIdx.x * blockDim.x + threadIdx.x;
    int i = idx & 15;
    int lane = (idx >> 4) & 31;
    int tilei = idx >> 9;
    int nkt = Kdim >> 5;
    int kt = tilei % nkt;
    int ntile = tilei / nkt;
    int n = ntile * 16 + (lane & 15);
    int k = kt * 32 + ((i < 8) ? (lane >> 4) * 8 + i : 16 + (lane >> 4) * 8 + (i - 8));
    dst[idx] = (bf16)W[(size_t)n * ld + coloff + k];
}

__global__ void gislow_kernel(const float* __restrict__ e, const float* __restrict__ wih,
                              const float* __restrict__ bih, float* __restrict__ out) {
    int idx = blockIdx.x * blockDim.x + threadIdx.x;  // B * 1536
    int b = idx / (3 * HS), j = idx % (3 * HS);
    float s = bih[j];
    const float* er = e + (size_t)b * ED;
    const float* wr = wih + (size_t)j * ED;
    for (int k = 0; k < ED; ++k) s += er[k] * wr[k];
    out[idx] = s;
}

__global__ void t0_kernel(const float* __restrict__ z, float* __restrict__ out) {
    int idx = blockIdx.x * blockDim.x + threadIdx.x;  // B*ZD
    size_t o = (size_t)idx * TT;
    out[o] = z[o];
    out[(size_t)BB * ZD * TT + o] = 0.0f;
}

// ---------------------------------------------------------------------------
// Main persistent kernel: grid = 16 blocks (batch tiles), block = 256 threads.
// ---------------------------------------------------------------------------
__launch_bounds__(256, 1)
__global__ void decoder_kernel(const float* __restrict__ z,
                               const float* __restrict__ noise,
                               const int* __restrict__ tfm,
                               const float* __restrict__ gis,    // [B][1536]
                               const float* __restrict__ sbhh,
                               const float* __restrict__ fbih,
                               const float* __restrict__ fbhh,
                               const float* __restrict__ mub,
                               const float* __restrict__ lvb,
                               const bf16* __restrict__ pWhh,
                               const bf16* __restrict__ pWz,
                               const bf16* __restrict__ pWh,
                               const bf16* __restrict__ pShh,
                               const bf16* __restrict__ pMu,
                               const bf16* __restrict__ pLv,
                               float* __restrict__ gihsAll,
                               float* __restrict__ out) {
    __shared__ __align__(16) bf16 h_rm[16 * HF];   // h_fast row-major
    __shared__ __align__(16) bf16 hA[16 * HF];     // h_fast A-fragment layout
    __shared__ __align__(16) bf16 hs_rm[16 * HS];  // h_slow row-major
    __shared__ __align__(16) bf16 hsA[16 * HS];    // h_slow A-fragment layout
    __shared__ __align__(16) bf16 z_rm[16 * ZD];   // z_prev row-major
    __shared__ __align__(16) bf16 zA[16 * ZD];     // z_prev A-fragment layout

    const int tid = threadIdx.x;
    const int lane = tid & 31;
    const int wave = tid >> 5;
    const int halfid = lane >> 4;
    const int ncol = lane & 15;
    const int b0 = blockIdx.x * 16;
    float* gihs = gihsAll + (size_t)blockIdx.x * (3 * 16 * HF);

    // ---- init state ----
    for (int i = tid; i < 16 * HF; i += 256) { h_rm[i] = (bf16)0.0f; hA[i] = (bf16)0.0f; }
    for (int i = tid; i < 16 * HS; i += 256) { hs_rm[i] = (bf16)0.0f; hsA[i] = (bf16)0.0f; }
    for (int i = tid; i < 16 * ZD; i += 256) {
        int M = i / ZD, c = i % ZD;
        z_rm[i] = (bf16)z[((size_t)(b0 + M) * ZD + c) * TT + 0];
    }
    __syncthreads();
    packA_lds(z_rm, zA, ZD, tid);
    __syncthreads();
    compute_gihs(hsA, pWh, fbih, fbhh, gihs, wave, lane, halfid, ncol);
    __syncthreads();

    // ---- sequential time loop ----
#pragma unroll 1
    for (int t = 1; t < TT; ++t) {
        // ---------- slow GRU (every K=4 steps, before fast uses it) ----------
        if ((t & 3) == 0) {
#pragma unroll 1
            for (int ct = 0; ct < 4; ++ct) {        // wave owns 64 h_slow cols
                int j0 = wave * 64 + ct * 16;
                int jc = j0 + ncol;
                v8f ar, az, ain, ahn;
                float bhr = sbhh[jc], bhz = sbhh[HS + jc], bhn = sbhh[2 * HS + jc];
#pragma unroll
                for (int v = 0; v < 8; ++v) {
                    int M = v + 8 * halfid;
                    const float* g = gis + (size_t)(b0 + M) * (3 * HS);
                    ar[v] = g[jc] + bhr;
                    az[v] = g[HS + jc] + bhz;
                    ain[v] = g[2 * HS + jc];
                    ahn[v] = bhn;
                }
                int nt = j0 >> 4;
#pragma unroll
                for (int kt = 0; kt < 16; ++kt) {
                    B3 c = loadB3(pShh, nt, 32 + nt, 64 + nt, 16, kt, lane);
                    v16bf a = load_frag(hsA + ((size_t)(kt * 32 + lane) << 4));
                    ar  = wmma_bf16(a, c.r, ar);
                    az  = wmma_bf16(a, c.z, az);
                    ahn = wmma_bf16(a, c.h, ahn);
                }
                sched_interleave<16, 6, 3, 27>();
#pragma unroll
                for (int v = 0; v < 8; ++v) {
                    int M = v + 8 * halfid;
                    float r = sigf(ar[v]), zg = sigf(az[v]);
                    float n = tanhf(ain[v] + r * ahn[v]);
                    float ho = (float)hs_rm[M * HS + jc];
                    hs_rm[M * HS + jc] = (bf16)((1.0f - zg) * n + zg * ho);
                }
            }
            __syncthreads();
            packA_lds(hs_rm, hsA, HS, tid);
            __syncthreads();
            compute_gihs(hsA, pWh, fbih, fbhh, gihs, wave, lane, halfid, ncol);
            __syncthreads();
        }

        // ---------- fast GRU: wave owns h_fast cols [wave*128, wave*128+128) ----
#pragma unroll 1
        for (int ct = 0; ct < 8; ++ct) {
            int j0 = wave * 128 + ct * 16;
            int jc = j0 + ncol;
            v8f ar, az, ain, ahn;
            float bhn = fbhh[2 * HF + jc];
#pragma unroll
            for (int v = 0; v < 8; ++v) {
                int M = v + 8 * halfid;
                ar[v]  = gihs[(size_t)(0 * 16 + M) * HF + jc];
                az[v]  = gihs[(size_t)(1 * 16 + M) * HF + jc];
                ain[v] = gihs[(size_t)(2 * 16 + M) * HF + jc];
                ahn[v] = bhn;
            }
            int nt = j0 >> 4;
            // z_prev contribution (K = 128)
#pragma unroll
            for (int kt = 0; kt < 4; ++kt) {
                B3 c = loadB3(pWz, nt, 64 + nt, 128 + nt, 4, kt, lane);
                v16bf a = load_frag(zA + ((size_t)(kt * 32 + lane) << 4));
                ar  = wmma_bf16(a, c.r, ar);
                az  = wmma_bf16(a, c.z, az);
                ain = wmma_bf16(a, c.h, ain);
            }
            // h_fast contribution (K = 1024)
#pragma unroll
            for (int kt = 0; kt < 32; ++kt) {
                B3 c = loadB3(pWhh, nt, 64 + nt, 128 + nt, 32, kt, lane);
                v16bf a = load_frag(hA + ((size_t)(kt * 32 + lane) << 4));
                ar  = wmma_bf16(a, c.r, ar);
                az  = wmma_bf16(a, c.z, az);
                ahn = wmma_bf16(a, c.h, ahn);
            }
            // combined pipeline spec over z-part + h-part (36 fused iterations),
            // with the ~25 accumulator-init loads absorbed in the prologue group.
            sched_interleave<36, 6, 3, 25>();
#pragma unroll
            for (int v = 0; v < 8; ++v) {
                int M = v + 8 * halfid;
                float r = sigf(ar[v]), zg = sigf(az[v]);
                float n = tanhf(ain[v] + r * ahn[v]);
                float ho = (float)h_rm[M * HF + jc];
                h_rm[M * HF + jc] = (bf16)((1.0f - zg) * n + zg * ho);
            }
        }
        __syncthreads();
        packA_lds(h_rm, hA, HF, tid);
        __syncthreads();

        // ---------- mu / lv heads + sampling: wave owns z-dim cols [16w,16w+16) ----
        {
            int jc = wave * 16 + ncol;
            v8f amu, alv;
            float mb = mub[jc], lb = lvb[jc];
#pragma unroll
            for (int v = 0; v < 8; ++v) { amu[v] = mb; alv[v] = lb; }
#pragma unroll
            for (int kt = 0; kt < 32; ++kt) {
                v16bf bm = load_frag(btile(pMu, wave, 32, kt, lane));
                v16bf bl = load_frag(btile(pLv, wave, 32, kt, lane));
                v16bf a = load_frag(hA + ((size_t)(kt * 32 + lane) << 4));
                amu = wmma_bf16(a, bm, amu);
                alv = wmma_bf16(a, bl, alv);
            }
            sched_interleave<32, 4, 2, 2>();
            int tv = tfm[t];
#pragma unroll
            for (int v = 0; v < 8; ++v) {
                int M = v + 8 * halfid;
                size_t o = ((size_t)(b0 + M) * ZD + jc) * TT + t;
                float m = amu[v], l = alv[v];
                out[o] = m;
                out[(size_t)BB * ZD * TT + o] = l;
                float zn = (tv == 1) ? z[o] : (m + __expf(0.5f * l) * noise[o]);
                z_rm[M * ZD + jc] = (bf16)zn;
            }
        }
        __syncthreads();
        packA_lds(z_rm, zA, ZD, tid);
        __syncthreads();
    }
}

// ---------------------------------------------------------------------------
extern "C" void kernel_launch(void* const* d_in, const int* in_sizes, int n_in,
                              void* d_out, int out_size, void* d_ws, size_t ws_size,
                              hipStream_t stream) {
    const float* z       = (const float*)d_in[0];
    const float* e       = (const float*)d_in[1];
    const float* noise   = (const float*)d_in[2];
    const int*   tfmask  = (const int*)d_in[3];
    const float* swih    = (const float*)d_in[4];
    const float* swhh    = (const float*)d_in[5];
    const float* sbih    = (const float*)d_in[6];
    const float* sbhh    = (const float*)d_in[7];
    const float* fwih    = (const float*)d_in[8];
    const float* fwhh    = (const float*)d_in[9];
    const float* fbih    = (const float*)d_in[10];
    const float* fbhh    = (const float*)d_in[11];
    const float* muw     = (const float*)d_in[12];
    const float* mub     = (const float*)d_in[13];
    const float* lvw     = (const float*)d_in[14];
    const float* lvb     = (const float*)d_in[15];
    float* out = (float*)d_out;

    char* ws = (char*)d_ws;
    bf16* pWhh = (bf16*)(ws + O_WHH);
    bf16* pWz  = (bf16*)(ws + O_WZ);
    bf16* pWh  = (bf16*)(ws + O_WH);
    bf16* pShh = (bf16*)(ws + O_SHH);
    bf16* pMu  = (bf16*)(ws + O_MUW);
    bf16* pLv  = (bf16*)(ws + O_LVW);
    float* gis  = (float*)(ws + O_GIS);
    float* gihs = (float*)(ws + O_GIHS);

    // --- prep: pack weights (bf16 WMMA B-fragment layout) ---
    pack_bfrag_kernel<<<(3072 * 1024) / 256, 256, 0, stream>>>(fwhh, pWhh, 1024, 1024, 0);
    pack_bfrag_kernel<<<(3072 * 128) / 256, 256, 0, stream>>>(fwih, pWz, 128, 640, 0);
    pack_bfrag_kernel<<<(3072 * 512) / 256, 256, 0, stream>>>(fwih, pWh, 512, 640, 128);
    pack_bfrag_kernel<<<(1536 * 512) / 256, 256, 0, stream>>>(swhh, pShh, 512, 512, 0);
    pack_bfrag_kernel<<<(128 * 1024) / 256, 256, 0, stream>>>(muw, pMu, 1024, 1024, 0);
    pack_bfrag_kernel<<<(128 * 1024) / 256, 256, 0, stream>>>(lvw, pLv, 1024, 1024, 0);

    // --- prep: time-invariant slow-GRU input gates, t=0 outputs ---
    gislow_kernel<<<(BB * 3 * HS) / 256, 256, 0, stream>>>(e, swih, sbih, gis);
    t0_kernel<<<(BB * ZD) / 256, 256, 0, stream>>>(z, out);

    // --- main persistent kernel: 16 batch tiles x 8 waves ---
    decoder_kernel<<<BB / 16, 256, 0, stream>>>(z, noise, tfmask, gis, sbhh, fbih, fbhh,
                                                mub, lvb, pWhh, pWz, pWh, pShh, pMu, pLv,
                                                gihs, out);
}
